// DQNet_90658169684065
// MI455X (gfx1250) — compile-verified, matrix operand order checked by reading, not run on hardware
//
#include <hip/hip_runtime.h>
#include <hip/hip_bf16.h>

typedef __attribute__((ext_vector_type(16))) _Float16 v16h;
typedef __attribute__((ext_vector_type(8)))  float    v8f;

namespace {
constexpr int NB   = 64;              // groups (B)
constexpr int NPG  = 100;             // nodes per group
constexpr int H    = 64;              // hidden
constexpr int KCLS = 5;               // label classes
constexpr int NA   = 50;              // actions per group
constexpr int EPG  = NPG * (NPG - 1); // 9900 edges per group
constexpr int MP   = 112;             // node rows padded to 16
constexpr int KP   = 128;             // K (source-node) dim padded to 32
}

// ---- WMMA f16 (16x16x32, f32 accum) fragment helpers -----------------------
// A (16x32 f16): lane l; M = m0 + (l&15); K-half kb = (l>>4)*8.
//   VGPR 0..3 hold K = kb+0..7, VGPR 4..7 hold K = 16+kb+0..7 (packed pairs).
// B uses the same per-lane pattern with M->N, so B stored transposed ([N][K])
// makes the B fragment identical to frag_A: two contiguous b128 loads per lane.
// C/D (16x16 f32): VGPR r holds row M = m0 + (l>>4)*8 + r, col N = n0 + (l&15).
__device__ inline v16h frag_A(const _Float16* p, int ld, int m0, int k0) {
  int lane = threadIdx.x & 31;
  const _Float16* row = p + (m0 + (lane & 15)) * ld + k0 + ((lane >> 4) << 3);
  v16h a;
#pragma unroll
  for (int j = 0; j < 8; ++j) {
    a[j]     = row[j];       // contiguous 8 halves  -> b128
    a[8 + j] = row[16 + j];  // contiguous 8 halves  -> b128
  }
  return a;
}

__device__ inline v8f wmma_f16(v16h a, v16h b, v8f c) {
  return __builtin_amdgcn_wmma_f32_16x16x32_f16(false, a, false, b, (short)0, c,
                                                false, false);
}

// ---- Kernel 1: edge prep --------------------------------------------------
__global__ void k_edge(const float* __restrict__ e_type,
                       const float* __restrict__ dvals,
                       const int* __restrict__ src, const int* __restrict__ dst,
                       const float* __restrict__ t4_w,
                       const float* __restrict__ t4_b,
                       _Float16* __restrict__ Wt_all,
                       float* __restrict__ t4s_all) {
  const int b = blockIdx.x, tid = threadIdx.x, bs = blockDim.x;
  __shared__ _Float16 sloc[EPG];  // 19.8 KB: s_e for this group's edges
  _Float16* Wt = Wt_all + (size_t)b * MP * KP;

  for (int idx = tid; idx < MP * KP; idx += bs) Wt[idx] = (_Float16)0.f;
  __syncthreads();

  const int e0 = b * EPG;
  for (int e = tid; e < EPG; e += bs) {
    const int ge = e0 + e;
    const float e1 = e_type[2 * ge];
    const float dd = dvals[ge];
    const int i = src[ge] - b * NPG;
    const int j = dst[ge] - b * NPG;
    Wt[j * KP + i] = (_Float16)(e1 * e1 * dd);
    sloc[e] = (_Float16)(dd * e1);
  }
  __syncthreads();

  // t4_sum[j][hh] = sum_i relu(s(i->j)*t4_w[hh]+t4_b[hh]); dense edge index
  float* t4s = t4s_all + (size_t)b * MP * H;
  for (int idx = tid; idx < MP * H; idx += bs) {
    const int j = idx >> 6, hh = idx & 63;
    float acc = 0.f;
    if (j < NPG) {
      const float w4 = t4_w[hh], b4 = t4_b[hh];
      for (int i = 0; i < NPG; ++i) {
        if (i == j) continue;
        const int e = i * (NPG - 1) + j - (j > i ? 1 : 0);
        acc += fmaxf(fmaf((float)sloc[e], w4, b4), 0.f);
      }
    }
    t4s[idx] = acc;
  }
}

// ---- Kernel 2: base = label@l1 + l1_b + t4_sum@t3_w + t3_b; h0 = relu(base+l2_b)
__global__ void k_base(const float* __restrict__ label,
                       const float* __restrict__ l1_w, const float* __restrict__ l1_b,
                       const float* __restrict__ t3_w, const float* __restrict__ t3_b,
                       const float* __restrict__ l2_b,
                       const float* __restrict__ t4s_all,
                       _Float16* __restrict__ base_all,
                       _Float16* __restrict__ hf_all) {
  const int b = blockIdx.x, tid = threadIdx.x, bs = blockDim.x;
  const int wid = tid >> 5, lane = tid & 31, nw = bs >> 5;
  __shared__ _Float16 t4f[MP * H];      // 14.3 KB (A operand, [M][K])
  __shared__ _Float16 w3T[H * H];       //  8.0 KB (B transposed, [N][K])
  __shared__ float    labl[NPG * KCLS]; //  2.0 KB (group label block)

  const float* t4s = t4s_all + (size_t)b * MP * H;
  _Float16* basef = base_all + (size_t)b * MP * H;
  _Float16* hfg   = hf_all + (size_t)b * KP * H;

  __builtin_prefetch(t3_w, 0, 3);
  for (int idx = tid; idx < MP * H; idx += bs) t4f[idx] = (_Float16)t4s[idx];
  for (int idx = tid; idx < H * H; idx += bs)   // coalesced read, LDS scatter
    w3T[(idx & 63) * H + (idx >> 6)] = (_Float16)t3_w[idx];
  for (int idx = tid; idx < NPG * KCLS; idx += bs)
    labl[idx] = label[(size_t)b * NPG * KCLS + idx];
  __syncthreads();

  // double tiles: 2 N-tiles per iteration, shared A frag, 2 accumulators
  const int ND = (MP / 16) * 2;  // 14
  for (int t = wid; t < ND; t += nw) {
    const int m0 = (t >> 1) * 16, n0 = (t & 1) * 32;
    v8f acc0 = {}, acc1 = {};
#pragma unroll
    for (int k0 = 0; k0 < H; k0 += 32) {
      const v16h a = frag_A(t4f, H, m0, k0);
      acc0 = wmma_f16(a, frag_A(w3T, H, n0, k0),      acc0);
      acc1 = wmma_f16(a, frag_A(w3T, H, n0 + 16, k0), acc1);
    }
    const int mb = m0 + ((lane >> 4) << 3);
#pragma unroll
    for (int half = 0; half < 2; ++half) {
      const v8f acc = half ? acc1 : acc0;
      const int n = n0 + half * 16 + (lane & 15);
#pragma unroll
      for (int r = 0; r < 8; ++r) {
        const int m = mb + r;
        float v = acc[r], h0 = 0.f;
        if (m < NPG) {
          const float* lr = labl + m * KCLS;
          float lab = l1_b[n] + t3_b[n];
#pragma unroll
          for (int k = 0; k < KCLS; ++k) lab = fmaf(lr[k], l1_w[k * H + n], lab);
          v += lab;
          h0 = fmaxf(v + l2_b[n], 0.f);  // first GNN step folded (h starts at 0)
        } else {
          v = 0.f;
        }
        basef[m * H + n] = (_Float16)v;
        hfg[m * H + n]   = (_Float16)h0;
      }
    }
  }
  // zero K-pad rows 112..127 of h (source dimension of Wt@h)
  for (int idx = tid; idx < (KP - MP) * H; idx += bs)
    hfg[MP * H + idx] = (_Float16)0.f;
}

// ---- Kernel 3: one GNN step: h = relu(base + (Wt@h)@l2_w + l2_b) ----------
__global__ void k_gnn(const _Float16* __restrict__ Wt_all,
                      const _Float16* __restrict__ base_all,
                      _Float16* __restrict__ hf_all,
                      const float* __restrict__ l2_w,
                      const float* __restrict__ l2_b) {
  const int b = blockIdx.x, tid = threadIdx.x, bs = blockDim.x;
  const int wid = tid >> 5, lane = tid & 31, nw = bs >> 5;
  __shared__ _Float16 hT[H * KP];   // 16.0 KB (staged h transposed, [N=H][K=i])
  __shared__ _Float16 w2T[H * H];   //  8.0 KB (l2_w transposed, [N][K])
  __shared__ _Float16 n1h[MP * H];  // 14.3 KB (Wt@h result, A operand [M][K])

  const _Float16* Wt = Wt_all + (size_t)b * MP * KP;
  const _Float16* basef = base_all + (size_t)b * MP * H;
  _Float16* hfg = hf_all + (size_t)b * KP * H;

  __builtin_prefetch(Wt, 0, 3);
  // coalesced global reads, transposed LDS scatter stores
  for (int idx = tid; idx < KP * H; idx += bs) {
    const _Float16 v = hfg[idx];                 // i = idx>>6, n = idx&63
    hT[(idx & 63) * KP + (idx >> 6)] = v;
  }
  for (int idx = tid; idx < H * H; idx += bs)
    w2T[(idx & 63) * H + (idx >> 6)] = (_Float16)l2_w[idx];
  __syncthreads();

  const int ND = (MP / 16) * 2;  // 14 double tiles
  // GEMM 1: n1 = Wt (112x128) @ h (128x64)   (A from global, B from LDS)
  for (int t = wid; t < ND; t += nw) {
    const int m0 = (t >> 1) * 16, n0 = (t & 1) * 32;
    v8f acc0 = {}, acc1 = {};
#pragma unroll
    for (int k0 = 0; k0 < KP; k0 += 32) {
      const v16h a = frag_A(Wt, KP, m0, k0);
      acc0 = wmma_f16(a, frag_A(hT, KP, n0, k0),      acc0);
      acc1 = wmma_f16(a, frag_A(hT, KP, n0 + 16, k0), acc1);
    }
    const int mb = m0 + ((lane >> 4) << 3);
#pragma unroll
    for (int half = 0; half < 2; ++half) {
      const v8f acc = half ? acc1 : acc0;
      const int n = n0 + half * 16 + (lane & 15);
#pragma unroll
      for (int r = 0; r < 8; ++r) n1h[(mb + r) * H + n] = (_Float16)acc[r];
    }
  }
  __syncthreads();
  // GEMM 2: h' = relu(base + n1 @ l2_w + l2_b)
  for (int t = wid; t < ND; t += nw) {
    const int m0 = (t >> 1) * 16, n0 = (t & 1) * 32;
    v8f acc0 = {}, acc1 = {};
#pragma unroll
    for (int k0 = 0; k0 < H; k0 += 32) {
      const v16h a = frag_A(n1h, H, m0, k0);
      acc0 = wmma_f16(a, frag_A(w2T, H, n0, k0),      acc0);
      acc1 = wmma_f16(a, frag_A(w2T, H, n0 + 16, k0), acc1);
    }
    const int mb = m0 + ((lane >> 4) << 3);
#pragma unroll
    for (int half = 0; half < 2; ++half) {
      const v8f acc = half ? acc1 : acc0;
      const int n = n0 + half * 16 + (lane & 15);
#pragma unroll
      for (int r = 0; r < 8; ++r) {
        const int m = mb + r;
        float v = fmaxf((float)basef[m * H + n] + acc[r] + l2_b[n], 0.f);
        hfg[m * H + n] = (_Float16)((m < NPG) ? v : 0.f);
      }
    }
  }
}

// ---- Kernel 4: dueling head ----------------------------------------------
__global__ void k_head(const _Float16* __restrict__ hf_all,
                       const float* __restrict__ t6_w,  const float* __restrict__ t6_b,
                       const float* __restrict__ t7_1_w,const float* __restrict__ t7_1_b,
                       const float* __restrict__ t7_2_w,const float* __restrict__ t7_2_b,
                       const float* __restrict__ t9_1_w,const float* __restrict__ t9_1_b,
                       const float* __restrict__ t9_2_w,const float* __restrict__ t9_2_b,
                       const float* __restrict__ t5_w,  const float* __restrict__ t5_b,
                       const int* __restrict__ actions,
                       float* __restrict__ out) {
  const int b = blockIdx.x, tid = threadIdx.x, bs = blockDim.x;
  const int wid = tid >> 5, lane = tid & 31, nw = bs >> 5;
  __shared__ _Float16 h2f[MP * H];      // 14.3 KB ([node][H])
  __shared__ _Float16 wT[H * H];        //  8.0 KB (B transposed: t7_1 -> t7_2 -> t9_2)
  __shared__ _Float16 haf[64 * H];      //  8.0 KB (relu(h2[a0]+h2[a1]), A operand)
  __shared__ _Float16 u1f[64 * H];      //  8.0 KB (A operand)
  __shared__ float    qbuf[64 * H];     // 16.0 KB
  __shared__ float    meanv[H], sm1[H], gvec[H];

  const _Float16* hfg = hf_all + (size_t)b * KP * H;
  __builtin_prefetch(t7_1_w, 0, 3);

  // mean over nodes, then stem / g_branch matvecs
  if (tid < H) {
    float s = 0.f;
    for (int i = 0; i < NPG; ++i) s += (float)hfg[i * H + tid];
    meanv[tid] = s * (1.f / NPG);
  }
  for (int idx = tid; idx < H * H; idx += bs)
    wT[(idx & 63) * H + (idx >> 6)] = (_Float16)t7_1_w[idx];
  __syncthreads();
  if (tid < H) {
    float s = t6_b[tid];
    for (int k = 0; k < H; ++k) s = fmaf(meanv[k], t6_w[k * H + tid], s);
    sm1[tid] = fmaxf(s, 0.f);
  }
  __syncthreads();
  if (tid < H) {
    float s = t9_1_b[tid];
    for (int k = 0; k < H; ++k) s = fmaf(sm1[k], t9_1_w[k * H + tid], s);
    gvec[tid] = s;
  }

  // h2 = h @ t7_1_w + t7_1_b   (112x64x64); A fragments straight from global
  const int ND1 = (MP / 16) * 2;  // 14
  for (int t = wid; t < ND1; t += nw) {
    const int m0 = (t >> 1) * 16, n0 = (t & 1) * 32;
    v8f acc0 = {}, acc1 = {};
#pragma unroll
    for (int k0 = 0; k0 < H; k0 += 32) {
      const v16h a = frag_A(hfg, H, m0, k0);
      acc0 = wmma_f16(a, frag_A(wT, H, n0, k0),      acc0);
      acc1 = wmma_f16(a, frag_A(wT, H, n0 + 16, k0), acc1);
    }
    const int mb = m0 + ((lane >> 4) << 3);
#pragma unroll
    for (int half = 0; half < 2; ++half) {
      const v8f acc = half ? acc1 : acc0;
      const int n = n0 + half * 16 + (lane & 15);
#pragma unroll
      for (int r = 0; r < 8; ++r)
        h2f[(mb + r) * H + n] = (_Float16)(acc[r] + t7_1_b[n]);
    }
  }
  __syncthreads();

  // gather actions: haf[a][:] = relu(h2[a0] + h2[a1]); reload wT with t7_2
  for (int idx = tid; idx < 64 * H; idx += bs) {
    const int row = idx >> 6, n = idx & 63;
    float v = 0.f;
    if (row < NA) {
      const int a0 = actions[(b * NA + row) * 2 + 0];
      const int a1 = actions[(b * NA + row) * 2 + 1];
      v = fmaxf((float)h2f[a0 * H + n] + (float)h2f[a1 * H + n], 0.f);
    }
    haf[idx] = (_Float16)v;
  }
  for (int idx = tid; idx < H * H; idx += bs)
    wT[(idx & 63) * H + (idx >> 6)] = (_Float16)t7_2_w[idx];
  __syncthreads();

  // u1 = relu(haf @ t7_2_w + t7_2_b)   (64x64x64)
  const int ND2 = (64 / 16) * 2;  // 8 double tiles
  for (int t = wid; t < ND2; t += nw) {
    const int m0 = (t >> 1) * 16, n0 = (t & 1) * 32;
    v8f acc0 = {}, acc1 = {};
#pragma unroll
    for (int k0 = 0; k0 < H; k0 += 32) {
      const v16h a = frag_A(haf, H, m0, k0);
      acc0 = wmma_f16(a, frag_A(wT, H, n0, k0),      acc0);
      acc1 = wmma_f16(a, frag_A(wT, H, n0 + 16, k0), acc1);
    }
    const int mb = m0 + ((lane >> 4) << 3);
#pragma unroll
    for (int half = 0; half < 2; ++half) {
      const v8f acc = half ? acc1 : acc0;
      const int n = n0 + half * 16 + (lane & 15);
#pragma unroll
      for (int r = 0; r < 8; ++r)
        u1f[(mb + r) * H + n] = (_Float16)fmaxf(acc[r] + t7_2_b[n], 0.f);
    }
  }
  __syncthreads();
  for (int idx = tid; idx < H * H; idx += bs)
    wT[(idx & 63) * H + (idx >> 6)] = (_Float16)t9_2_w[idx];
  __syncthreads();

  // q = relu(g[n] + u1 @ t9_2_w + t9_2_b)
  for (int t = wid; t < ND2; t += nw) {
    const int m0 = (t >> 1) * 16, n0 = (t & 1) * 32;
    v8f acc0 = {}, acc1 = {};
#pragma unroll
    for (int k0 = 0; k0 < H; k0 += 32) {
      const v16h a = frag_A(u1f, H, m0, k0);
      acc0 = wmma_f16(a, frag_A(wT, H, n0, k0),      acc0);
      acc1 = wmma_f16(a, frag_A(wT, H, n0 + 16, k0), acc1);
    }
    const int mb = m0 + ((lane >> 4) << 3);
#pragma unroll
    for (int half = 0; half < 2; ++half) {
      const v8f acc = half ? acc1 : acc0;
      const int n = n0 + half * 16 + (lane & 15);
#pragma unroll
      for (int r = 0; r < 8; ++r)
        qbuf[(mb + r) * H + n] = fmaxf(gvec[n] + acc[r] + t9_2_b[n], 0.f);
    }
  }
  __syncthreads();

  // Q_sa[m] = qbuf[m] . t5_w + t5_b
  for (int m = tid; m < NA; m += bs) {
    float s = t5_b[0];
    for (int n = 0; n < H; ++n) s = fmaf(qbuf[m * H + n], t5_w[n], s);
    out[b * NA + m] = s;
  }
}

extern "C" void kernel_launch(void* const* d_in, const int* in_sizes, int n_in,
                              void* d_out, int out_size, void* d_ws, size_t ws_size,
                              hipStream_t stream) {
  const float* label  = (const float*)d_in[0];
  const float* e_type = (const float*)d_in[1];
  const float* dvals  = (const float*)d_in[2];
  const float* l1_w   = (const float*)d_in[3];
  const float* l1_b   = (const float*)d_in[4];
  const float* l2_w   = (const float*)d_in[5];
  const float* l2_b   = (const float*)d_in[6];
  const float* t3_w   = (const float*)d_in[7];
  const float* t3_b   = (const float*)d_in[8];
  const float* t4_w   = (const float*)d_in[9];
  const float* t4_b   = (const float*)d_in[10];
  const float* t5_w   = (const float*)d_in[11];
  const float* t5_b   = (const float*)d_in[12];
  const float* t6_w   = (const float*)d_in[13];
  const float* t6_b   = (const float*)d_in[14];
  const float* t7_1_w = (const float*)d_in[15];
  const float* t7_1_b = (const float*)d_in[16];
  const float* t7_2_w = (const float*)d_in[17];
  const float* t7_2_b = (const float*)d_in[18];
  const float* t9_1_w = (const float*)d_in[19];
  const float* t9_1_b = (const float*)d_in[20];
  const float* t9_2_w = (const float*)d_in[21];
  const float* t9_2_b = (const float*)d_in[22];
  const int*   src    = (const int*)d_in[23];
  const int*   dst    = (const int*)d_in[24];
  const int*   acts   = (const int*)d_in[25];
  float* out = (float*)d_out;

  // workspace carve-up (all offsets 16B-aligned); total ~5.6 MB (L2 resident)
  char* ws = (char*)d_ws;
  size_t off = 0;
  _Float16* Wt_all   = (_Float16*)(ws + off); off += (size_t)NB * MP * KP * sizeof(_Float16);
  float*    t4s_all  = (float*)(ws + off);    off += (size_t)NB * MP * H  * sizeof(float);
  _Float16* base_all = (_Float16*)(ws + off); off += (size_t)NB * MP * H  * sizeof(_Float16);
  _Float16* hf_all   = (_Float16*)(ws + off); off += (size_t)NB * KP * H  * sizeof(_Float16);
  (void)off; (void)ws_size; (void)in_sizes; (void)n_in; (void)out_size;

  k_edge<<<NB, 256, 0, stream>>>(e_type, dvals, src, dst, t4_w, t4_b, Wt_all, t4s_all);
  k_base<<<NB, 128, 0, stream>>>(label, l1_w, l1_b, t3_w, t3_b, l2_b, t4s_all,
                                 base_all, hf_all);
  // GNN steps 2 and 3 (step 1 folded into k_base since h starts at zero)
  k_gnn<<<NB, 128, 0, stream>>>(Wt_all, base_all, hf_all, l2_w, l2_b);
  k_gnn<<<NB, 128, 0, stream>>>(Wt_all, base_all, hf_all, l2_w, l2_b);
  k_head<<<NB, 128, 0, stream>>>(hf_all, t6_w, t6_b, t7_1_w, t7_1_b, t7_2_w, t7_2_b,
                                 t9_1_w, t9_1_b, t9_2_w, t9_2_b, t5_w, t5_b, acts, out);
}
